// LMModel_RNN_4406636446317
// MI455X (gfx1250) — matrix operand order repaired and checked by reading, not compiled, not run on hardware
//
#include <hip/hip_runtime.h>
#include <hip/hip_bf16.h>
#include <math.h>

// ---------------------------------------------------------------------------
// MI455X (gfx1250) implementation.
//
// Roofline: decoder GEMM [8192x256]x[256x32000] = 134 GFLOP, 1.05 GB f32
// output stores -> ~45us at 23.3 TB/s; dec_W (16 MB f16) is L2-resident
// (192 MB L2) so the kernel is store-bound. Use f16 inputs with f32
// accumulation via v_wmma_f32_16x16x32_f16 (wave32 WMMA, CDNA5).
// The RNN recurrence (8.6 GFLOP, sequential) runs on a single WGP with all
// state in LDS, also using WMMA.
// ---------------------------------------------------------------------------

typedef _Float16 half_t;
typedef __attribute__((ext_vector_type(16))) _Float16 v16h;
typedef __attribute__((ext_vector_type(8)))  _Float16 v8h;
typedef __attribute__((ext_vector_type(8)))  float    v8f;

#define NVOC    32000
#define DIM     256
#define HID     256
#define NLAYERS 4
#define TT      256
#define BB      32

__device__ __forceinline__ v8f wmma_f16(v16h a, v16h b, v8f c) {
  // D = A(16x32) * B(32x16) + C(16x16 f32)
  return __builtin_amdgcn_wmma_f32_16x16x32_f16(
      /*neg_a=*/false, a, /*neg_b=*/false, b,
      /*c_mod=*/(short)0, c, /*reuse_a=*/false, /*reuse_b=*/false);
}

// A fragment: A is [M,K] row-major f16. Lane holds row m = m0 + (lane&15),
// half = lane>>4. Element i of v16h: k = k0 + 16*half + i.
// => two contiguous 8-half (16B) chunks at k0+16h and k0+16h+8.
__device__ __forceinline__ v16h load_a_frag(const half_t* A, int lda,
                                            int m0, int k0, int lane) {
  const int m = m0 + (lane & 15);
  const int h = lane >> 4;
  const v8h lo = *(const v8h*)(A + (size_t)m * lda + k0 + 16 * h);
  const v8h hi = *(const v8h*)(A + (size_t)m * lda + k0 + 16 * h + 8);
  v16h r;
  for (int i = 0; i < 8; ++i) { r[i] = lo[i]; r[8 + i] = hi[i]; }
  return r;
}

// B fragment for out = X * W^T where W is [N,K] row-major f16 (B(k,n)=W[n,k]).
// Lane holds column n = n0 + (lane&15), half = lane>>4.
// Element i of v16h: k = k0 + 16*half + i  => one contiguous 32B v16h load.
__device__ __forceinline__ v16h load_b_frag(const half_t* W, int ldw,
                                            int n0, int k0, int lane) {
  const int n = n0 + (lane & 15);
  const int h = lane >> 4;
  return *(const v16h*)(W + (size_t)n * ldw + k0 + 16 * h);
}

// ---------------------------------------------------------------------------
// f32 -> f16 conversion
// ---------------------------------------------------------------------------
__global__ void f32_to_f16_kernel(const float* __restrict__ in,
                                  half_t* __restrict__ out, int n) {
  int i = blockIdx.x * blockDim.x + threadIdx.x;
  if (i < n) out[i] = (half_t)in[i];
}

// ---------------------------------------------------------------------------
// RNN recurrence: one workgroup (8 wave32), all state in LDS.
// h_new[l] = tanh(x @ W_ih[l]^T + b_ih[l] + h[l] @ W_hh[l]^T + b_hh[l])
// Writes top-layer outputs (f16) for the decoder, final hidden (f32) to d_out.
// ---------------------------------------------------------------------------
__global__ __launch_bounds__(256) void rnn_kernel(
    const int* __restrict__ tokens, const float* __restrict__ emb,
    const half_t* __restrict__ Wih, const half_t* __restrict__ Whh,
    const float* __restrict__ bih, const float* __restrict__ bhh,
    half_t* __restrict__ outputs_f16, float* __restrict__ hidden_out) {
  __shared__ __align__(16) half_t x_lds[BB][HID];           // 16 KB
  __shared__ __align__(16) half_t h_lds[NLAYERS][BB][HID];  // 64 KB

  const int tid  = threadIdx.x;
  const int wave = tid >> 5;
  const int lane = tid & 31;
  const int n0   = wave * 32;  // this wave owns columns [n0, n0+32)

  // zero-init carried hidden state
  for (int i = tid; i < NLAYERS * BB * HID; i += 256)
    ((half_t*)h_lds)[i] = (half_t)0.f;
  __syncthreads();

  for (int t = 0; t < TT; ++t) {
    // x = emb[tokens[t,:]]  (f32 -> f16 into LDS)
    for (int i = tid; i < BB * HID; i += 256) {
      const int b = i >> 8, d = i & 255;
      const int tok = tokens[t * BB + b];
      x_lds[b][d] = (half_t)emb[(size_t)tok * DIM + d];
    }
    __syncthreads();

    for (int l = 0; l < NLAYERS; ++l) {
      const half_t* Wi = Wih + (size_t)l * HID * HID;
      const half_t* Wh = Whh + (size_t)l * HID * HID;

      // seed accumulators with the (n-dependent) biases
      const float bias0 = bih[l * HID + n0 + (lane & 15)]
                        + bhh[l * HID + n0 + (lane & 15)];
      const float bias1 = bih[l * HID + n0 + 16 + (lane & 15)]
                        + bhh[l * HID + n0 + 16 + (lane & 15)];
      v8f acc[2][2];  // [m-tile][n-tile]
      for (int r = 0; r < 8; ++r) {
        acc[0][0][r] = bias0; acc[1][0][r] = bias0;
        acc[0][1][r] = bias1; acc[1][1][r] = bias1;
      }

      for (int kb = 0; kb < HID; kb += 32) {
        // x @ W_ih^T
        v16h a0 = load_a_frag(&x_lds[0][0], HID, 0,  kb, lane);
        v16h a1 = load_a_frag(&x_lds[0][0], HID, 16, kb, lane);
        v16h b0 = load_b_frag(Wi, HID, n0,      kb, lane);
        v16h b1 = load_b_frag(Wi, HID, n0 + 16, kb, lane);
        acc[0][0] = wmma_f16(a0, b0, acc[0][0]);
        acc[0][1] = wmma_f16(a0, b1, acc[0][1]);
        acc[1][0] = wmma_f16(a1, b0, acc[1][0]);
        acc[1][1] = wmma_f16(a1, b1, acc[1][1]);
        // h[l] @ W_hh^T
        v16h c0 = load_a_frag(&h_lds[l][0][0], HID, 0,  kb, lane);
        v16h c1 = load_a_frag(&h_lds[l][0][0], HID, 16, kb, lane);
        v16h d0 = load_b_frag(Wh, HID, n0,      kb, lane);
        v16h d1 = load_b_frag(Wh, HID, n0 + 16, kb, lane);
        acc[0][0] = wmma_f16(c0, d0, acc[0][0]);
        acc[0][1] = wmma_f16(c0, d1, acc[0][1]);
        acc[1][0] = wmma_f16(c1, d0, acc[1][0]);
        acc[1][1] = wmma_f16(c1, d1, acc[1][1]);
      }

      __syncthreads();  // all waves done READING x_lds / h_lds[l]

      // tanh, write back as next x and carried h[l]
      for (int mt = 0; mt < 2; ++mt) {
        for (int nt = 0; nt < 2; ++nt) {
          const int n = n0 + nt * 16 + (lane & 15);
          for (int r = 0; r < 8; ++r) {
            const int m = mt * 16 + (lane >> 4) * 8 + r;
            const half_t hv = (half_t)tanhf(acc[mt][nt][r]);
            h_lds[l][m][n] = hv;
            x_lds[m][n]    = hv;
            if (l == NLAYERS - 1)
              outputs_f16[((size_t)t * BB + m) * HID + n] = hv;
          }
        }
      }
      __syncthreads();  // writes visible before next layer reads
    }
  }

  // final hidden [L,B,H] in f32
  for (int i = tid; i < NLAYERS * BB * HID; i += 256)
    hidden_out[i] = (float)((half_t*)h_lds)[i];
}

// ---------------------------------------------------------------------------
// Decoder GEMM: out[m, v] = sum_h A[m,h] * W[v,h] + dec_b[v]
// M = T*B = 8192, K = 256, N = V = 32000.
// Block = 8 waves in a 2x4 grid -> 64(M) x 128(N) tile; 32x32 per wave.
// ---------------------------------------------------------------------------
__global__ __launch_bounds__(256) void decoder_kernel(
    const half_t* __restrict__ A,      // [M, 256] f16
    const half_t* __restrict__ W,      // [V, 256] f16 (L2-resident, 16 MB)
    const float* __restrict__ dec_b,   // [V] f32
    float* __restrict__ out) {         // [M, V] f32
  const int K = HID;
  const int V = NVOC;
  const int wave = threadIdx.x >> 5;
  const int lane = threadIdx.x & 31;

  const int m0 = blockIdx.y * 64 + (wave >> 2) * 32;
  const int n0 = blockIdx.x * 128 + (wave & 3) * 32;

  v8f acc[2][2] = {};  // zero f32 accumulators
  for (int kb = 0; kb < K; kb += 32) {
    v16h a0 = load_a_frag(A, K, m0,      kb, lane);
    v16h a1 = load_a_frag(A, K, m0 + 16, kb, lane);
    v16h b0 = load_b_frag(W, K, n0,      kb, lane);
    v16h b1 = load_b_frag(W, K, n0 + 16, kb, lane);
    acc[0][0] = wmma_f16(a0, b0, acc[0][0]);
    acc[0][1] = wmma_f16(a0, b1, acc[0][1]);
    acc[1][0] = wmma_f16(a1, b0, acc[1][0]);
    acc[1][1] = wmma_f16(a1, b1, acc[1][1]);
  }

  const int nA = n0 + (lane & 15);
  const int nB = nA + 16;
  const float dbA = dec_b[nA];
  const float dbB = dec_b[nB];
  for (int mt = 0; mt < 2; ++mt) {
    for (int r = 0; r < 8; ++r) {
      const int m = m0 + mt * 16 + (lane >> 4) * 8 + r;
      out[(size_t)m * V + nA] = acc[mt][0][r] + dbA;
      out[(size_t)m * V + nB] = acc[mt][1][r] + dbB;
    }
  }
}

// ---------------------------------------------------------------------------
// Host-side launcher
// ---------------------------------------------------------------------------
extern "C" void kernel_launch(void* const* d_in, const int* in_sizes, int n_in,
                              void* d_out, int out_size, void* d_ws,
                              size_t ws_size, hipStream_t stream) {
  (void)in_sizes; (void)n_in; (void)out_size; (void)ws_size;

  const int*   tokens = (const int*)d_in[0];
  const float* emb    = (const float*)d_in[1];
  const float* W_ih   = (const float*)d_in[2];
  const float* W_hh   = (const float*)d_in[3];
  const float* b_ih   = (const float*)d_in[4];
  const float* b_hh   = (const float*)d_in[5];
  const float* dec_W  = (const float*)d_in[6];
  const float* dec_b  = (const float*)d_in[7];

  float* out = (float*)d_out;                               // decoded [T,B,V]
  float* hidden_out = out + (size_t)TT * BB * NVOC;         // hidden [L,B,H]

  // Workspace layout (all 256B aligned):
  char* ws = (char*)d_ws;
  half_t* decW_f16    = (half_t*)(ws);                      // 16,384,000 B
  half_t* Wih_f16     = (half_t*)(ws + 16384000);           //    524,288 B
  half_t* Whh_f16     = (half_t*)(ws + 16908288);           //    524,288 B
  half_t* outputs_f16 = (half_t*)(ws + 17432576);           //  4,194,304 B

  const int n_decW = NVOC * HID;
  f32_to_f16_kernel<<<(n_decW + 255) / 256, 256, 0, stream>>>(dec_W, decW_f16,
                                                              n_decW);
  const int n_W = NLAYERS * HID * HID;
  f32_to_f16_kernel<<<(n_W + 255) / 256, 256, 0, stream>>>(W_ih, Wih_f16, n_W);
  f32_to_f16_kernel<<<(n_W + 255) / 256, 256, 0, stream>>>(W_hh, Whh_f16, n_W);

  rnn_kernel<<<1, 256, 0, stream>>>(tokens, emb, Wih_f16, Whh_f16, b_ih, b_hh,
                                    outputs_f16, hidden_out);

  dim3 grid(NVOC / 128, (TT * BB) / 64);  // 250 x 128
  decoder_kernel<<<grid, 256, 0, stream>>>(outputs_f16, decW_f16, dec_b, out);
}
